// _BiHBLSTMLayer_76416058130987
// MI455X (gfx1250) — matrix operand order repaired and checked by reading, not compiled, not run on hardware
//
#include <hip/hip_runtime.h>

typedef __attribute__((ext_vector_type(16))) __bf16        v16bf;
typedef __attribute__((ext_vector_type(8)))  float         v8f;
typedef __attribute__((ext_vector_type(8)))  unsigned int  v8u;

#define B_   64
#define T_   512
#define I_   512
#define H_   512
#define G_   5
#define GH   2560
#define MTOT 32768   // T_*B_

// workspace layout (bytes), all offsets 16B aligned
#define XP_OFF   0ull                         // 2*32768*2560 bf16 = 335544320 B
#define WHB_OFF  335544320ull                 // 2*2560*512 bf16   =   5242880 B
#define HB_OFF   340787200ull                 // 2 bufs * 2 dirs * 64*512 bf16 = 262144 B
#define K_OFF    341049344ull                 // 2*64*512 f32 = 262144 B
#define KP_OFF   341311488ull                 // 2*64*512 f32 = 262144 B

__device__ __forceinline__ unsigned short f2bf(float f) {
  unsigned int u = __builtin_bit_cast(unsigned int, f);
  u += 0x7FFFu + ((u >> 16) & 1u);            // round-to-nearest-even
  return (unsigned short)(u >> 16);
}
__device__ __forceinline__ float bf2f(unsigned short s) {
  unsigned int u = ((unsigned int)s) << 16;
  return __builtin_bit_cast(float, u);
}
__device__ __forceinline__ v16bf frag_from(uint4 a, uint4 b) {
  v8u u;
  u[0] = a.x; u[1] = a.y; u[2] = a.z; u[3] = a.w;
  u[4] = b.x; u[5] = b.y; u[6] = b.z; u[7] = b.w;
  return __builtin_bit_cast(v16bf, u);
}
// A fragment (16x32 bf16): lane&15 = row M; half0 K=0..15  half1 K=16..31
__device__ __forceinline__ v16bf load_fragA(const unsigned int* lds, int row, int hf) {
  uint4 a = *(const uint4*)(lds + row * 24 + hf * 8);
  uint4 b = *(const uint4*)(lds + row * 24 + hf * 8 + 4);
  return frag_from(a, b);
}
// B fragment (32x16 bf16): lane&15 = col N; half0 K=0..15  half1 K=16..31
__device__ __forceinline__ v16bf load_fragB(const unsigned int* lds, int nrow, int hf) {
  const uint4* p = (const uint4*)(lds + nrow * 24 + hf * 8);
  uint4 a = p[0], b = p[1];
  return frag_from(a, b);
}
__device__ __forceinline__ float sigmoidf_(float x) { return 1.0f / (1.0f + __expf(-x)); }

// ---------------------------------------------------------------------------
// Kernel 0: convert Wh (both dirs) fp32 -> bf16; zero h/k/kp state
// ---------------------------------------------------------------------------
__global__ void init_kernel(const float* __restrict__ Whf, const float* __restrict__ Whb,
                            unsigned short* __restrict__ whb16,
                            unsigned int* __restrict__ hbu,
                            float* __restrict__ kst, float* __restrict__ kp) {
  const int n = G_ * H_ * H_;                 // 1310720 per dir
  const int stride = gridDim.x * blockDim.x;
  for (int i = blockIdx.x * blockDim.x + threadIdx.x; i < n; i += stride) {
    whb16[i]     = f2bf(Whf[i]);
    whb16[n + i] = f2bf(Whb[i]);
  }
  for (int i = blockIdx.x * blockDim.x + threadIdx.x; i < 2 * B_ * H_; i += stride) {
    hbu[i] = 0u;                              // both h ping-pong buffers (bf16 pairs)
    kst[i] = 0.0f;
    kp[i]  = 0.0f;
  }
}

// ---------------------------------------------------------------------------
// Kernel 1: xp[dir][s*64+b][n] = (x . Wx^T + bx) in bf16.  M=32768 N=2560 K=512
// Block: 256 thr (8 waves), tile 128x64, each wave 2x2 WMMA tiles, K chunk 32.
// Staging stays synchronous: it performs fp32->bf16 conversion in VGPRs.
// ---------------------------------------------------------------------------
__global__ __launch_bounds__(256) void xp_gemm_kernel(
    const float* __restrict__ x,
    const float* __restrict__ Wxf, const float* __restrict__ bxf,
    const float* __restrict__ Wxb, const float* __restrict__ bxb,
    unsigned short* __restrict__ xp) {
  __shared__ __attribute__((aligned(16))) unsigned int ldsA[128 * 24];
  __shared__ __attribute__((aligned(16))) unsigned int ldsB[64 * 24];

  const int tid = threadIdx.x;
  const int lane = tid & 31, w = tid >> 5;
  const int l = lane & 15, hf = lane >> 4;
  const int mw = w & 3, nw = w >> 2;
  const int dir = blockIdx.z;
  const int Nbase = blockIdx.x * 64;
  const int Mbase = blockIdx.y * 128;
  const float* Wx = dir ? Wxb : Wxf;
  const float* bx = dir ? bxb : bxf;

  v8f zero = {0.f, 0.f, 0.f, 0.f, 0.f, 0.f, 0.f, 0.f};
  v8f c00 = zero, c01 = zero, c10 = zero, c11 = zero;

  for (int kc = 0; kc < 16; ++kc) {
    const int k0 = kc * 32;
    // stage A: 128 rows x 32 k (fp32 -> bf16)
#pragma unroll
    for (int i = 0; i < 4; ++i) {
      int idx = i * 256 + tid;
      int row = idx >> 3, kq = idx & 7;
      int rg = Mbase + row;
      int bb = rg & 63, ss = rg >> 6;
      int t = dir ? (T_ - 1 - ss) : ss;       // backward dir consumes reversed time
      float4 v = *(const float4*)(x + ((size_t)bb * T_ + t) * I_ + k0 + kq * 4);
      ldsA[row * 24 + kq * 2]     = (unsigned)f2bf(v.x) | ((unsigned)f2bf(v.y) << 16);
      ldsA[row * 24 + kq * 2 + 1] = (unsigned)f2bf(v.z) | ((unsigned)f2bf(v.w) << 16);
    }
    // stage B: 64 n-rows x 32 k (Wx is [n][k] row-major)
#pragma unroll
    for (int i = 0; i < 2; ++i) {
      int idx = i * 256 + tid;
      int row = idx >> 3, kq = idx & 7;
      float4 v = *(const float4*)(Wx + (size_t)(Nbase + row) * I_ + k0 + kq * 4);
      ldsB[row * 24 + kq * 2]     = (unsigned)f2bf(v.x) | ((unsigned)f2bf(v.y) << 16);
      ldsB[row * 24 + kq * 2 + 1] = (unsigned)f2bf(v.z) | ((unsigned)f2bf(v.w) << 16);
    }
    __syncthreads();
    v16bf a0 = load_fragA(ldsA, mw * 32 + l, hf);
    v16bf a1 = load_fragA(ldsA, mw * 32 + 16 + l, hf);
    v16bf b0 = load_fragB(ldsB, nw * 32 + l, hf);
    v16bf b1 = load_fragB(ldsB, nw * 32 + 16 + l, hf);
    c00 = __builtin_amdgcn_wmma_f32_16x16x32_bf16(false, a0, false, b0, (short)0, c00, false, false);
    c01 = __builtin_amdgcn_wmma_f32_16x16x32_bf16(false, a0, false, b1, (short)0, c01, false, false);
    c10 = __builtin_amdgcn_wmma_f32_16x16x32_bf16(false, a1, false, b0, (short)0, c10, false, false);
    c11 = __builtin_amdgcn_wmma_f32_16x16x32_bf16(false, a1, false, b1, (short)0, c11, false, false);
    __syncthreads();
  }

  const float bx0 = bx[Nbase + nw * 32 + l];
  const float bx1 = bx[Nbase + nw * 32 + 16 + l];
  const size_t xpo = (size_t)dir * MTOT * GH;
#pragma unroll
  for (int r = 0; r < 8; ++r) {
    int m0 = Mbase + mw * 32 + r + 8 * hf;    // C/D layout: row = r + 8*half
    int m1 = m0 + 16;
    int n0 = Nbase + nw * 32 + l;
    int n1 = n0 + 16;
    xp[xpo + (size_t)m0 * GH + n0] = f2bf(c00[r] + bx0);
    xp[xpo + (size_t)m0 * GH + n1] = f2bf(c01[r] + bx1);
    xp[xpo + (size_t)m1 * GH + n0] = f2bf(c10[r] + bx0);
    xp[xpo + (size_t)m1 * GH + n1] = f2bf(c11[r] + bx1);
  }
}

// ---------------------------------------------------------------------------
// Kernel 2 (one launch per timestep): g = h.Wh^T + xp_t + bh; gate update.
// Block: 128 thr (4 waves). blockIdx = (hh_tile 0..31, dir 0..1).
// h is already bf16-packed -> stage it with GLOBAL_LOAD_ASYNC_TO_LDS_B128
// (ASYNCcnt path, bypasses VGPRs, overlaps with WMMA issue).
// Each wave owns one 16-batch x 16-hidden slice for ALL 5 gates -> pointwise
// update fused in WMMA accumulator registers.
// ---------------------------------------------------------------------------
__global__ __launch_bounds__(128) void lstm_step_kernel(
    const unsigned short* __restrict__ xp,
    const unsigned short* __restrict__ whb,
    const float* __restrict__ bhf, const float* __restrict__ bhb,
    unsigned short* __restrict__ hb,
    float* __restrict__ kst, float* __restrict__ kp,
    float* __restrict__ out, int s) {
  __shared__ __attribute__((aligned(16))) unsigned int ldsH[64 * 24];

  const int tid = threadIdx.x;
  const int lane = tid & 31, w = tid >> 5;
  const int l = lane & 15, hf = lane >> 4;
  const int dir = blockIdx.y;
  const int hhb = blockIdx.x * 16;

  const unsigned short* hr = hb + (size_t)(s & 1) * 2 * B_ * H_ + (size_t)dir * B_ * H_;
  unsigned short* hw       = hb + (size_t)((s & 1) ^ 1) * 2 * B_ * H_ + (size_t)dir * B_ * H_;
  const unsigned short* wh = whb + (size_t)dir * GH * H_;

  v8f zero = {0.f, 0.f, 0.f, 0.f, 0.f, 0.f, 0.f, 0.f};
  v8f acc[5];
#pragma unroll
  for (int g = 0; g < 5; ++g) acc[g] = zero;

  for (int kc = 0; kc < 16; ++kc) {
    // stage h chunk: 64 batch rows x 32 k, bf16 -> async DMA to LDS (no VGPRs).
    // Low 32 bits of a generic LDS pointer are the LDS byte address.
#pragma unroll
    for (int i = 0; i < 2; ++i) {
      int idx = i * 128 + tid;
      int row = idx >> 2, q = idx & 3;
      unsigned lds_off = (unsigned)(size_t)(ldsH + row * 24 + q * 4);
      unsigned long long gaddr =
          (unsigned long long)(size_t)(hr + (size_t)row * H_ + kc * 32 + q * 8);
      asm volatile("global_load_async_to_lds_b128 %0, %1, off"
                   :: "v"(lds_off), "v"(gaddr)
                   : "memory");
    }
    // prefetch next K-chunk of Wh toward the WGP (L2-resident weights)
    if (kc < 15) {
      const unsigned short* pf = wh + (size_t)(hhb + l) * H_ + (kc + 1) * 32 + hf * 16;
      __builtin_prefetch(pf, 0, 0);
    }
    asm volatile("s_wait_asynccnt 0x0" ::: "memory");
    __syncthreads();
    v16bf a = load_fragA(ldsH, w * 16 + l, hf);
#pragma unroll
    for (int g = 0; g < 5; ++g) {
      // Wh row n = g*512 + hh + l, k contiguous: 32B per lane, L2-resident
      const unsigned short* bp = wh + (size_t)(g * H_ + hhb + l) * H_ + kc * 32 + hf * 16;
      uint4 b0 = *(const uint4*)bp;
      uint4 b1 = *(const uint4*)(bp + 8);
      v16bf bf = frag_from(b0, b1);
      acc[g] = __builtin_amdgcn_wmma_f32_16x16x32_bf16(false, a, false, bf, (short)0, acc[g], false, false);
    }
    __syncthreads();
  }

  const float* bh = dir ? bhb : bhf;
  float bhv[5];
#pragma unroll
  for (int g = 0; g < 5; ++g) bhv[g] = bh[g * H_ + hhb + l];

  const size_t xpo = (size_t)dir * MTOT * GH + (size_t)s * B_ * GH;
  float* kd  = kst + dir * B_ * H_;
  float* kpd = kp + dir * B_ * H_;
  const int t_out = dir ? (T_ - 1 - s) : s;

#pragma unroll
  for (int r = 0; r < 8; ++r) {
    int bb = w * 16 + r + 8 * hf;             // batch row for this accumulator slot
    size_t xrow = xpo + (size_t)bb * GH + hhb + l;
    float gf = acc[0][r] + bhv[0] + bf2f(xp[xrow + 0 * H_]);
    float gi = acc[1][r] + bhv[1] + bf2f(xp[xrow + 1 * H_]);
    float go = acc[2][r] + bhv[2] + bf2f(xp[xrow + 2 * H_]);
    float gu = acc[3][r] + bhv[3] + bf2f(xp[xrow + 3 * H_]);
    float gk = acc[4][r] + bhv[4] + bf2f(xp[xrow + 4 * H_]);
    float f  = sigmoidf_(gf);
    float ii = sigmoidf_(gi);
    float o  = sigmoidf_(go);
    float u  = sigmoidf_(gu);
    float kpre = tanhf(gk);
    int sidx = bb * H_ + hhb + l;
    float kpo = kpd[sidx], ko = kd[sidx];
    float kpn = u * kpre + (1.0f - u) * kpo;
    float kn  = f * ko + ii * kpn;
    float hn  = o * tanhf(kn);
    kpd[sidx] = kpn;
    kd[sidx]  = kn;
    hw[sidx]  = f2bf(hn);
    out[((size_t)bb * T_ + t_out) * (2 * H_) + dir * H_ + hhb + l] = hn;
  }
}

// ---------------------------------------------------------------------------
extern "C" void kernel_launch(void* const* d_in, const int* in_sizes, int n_in,
                              void* d_out, int out_size, void* d_ws, size_t ws_size,
                              hipStream_t stream) {
  const float* x   = (const float*)d_in[0];
  const float* Wxf = (const float*)d_in[1];
  const float* bxf = (const float*)d_in[2];
  const float* Whf = (const float*)d_in[3];
  const float* bhf = (const float*)d_in[4];
  const float* Wxb = (const float*)d_in[5];
  const float* bxb = (const float*)d_in[6];
  const float* Whb = (const float*)d_in[7];
  const float* bhb = (const float*)d_in[8];
  float* out = (float*)d_out;

  char* ws = (char*)d_ws;
  unsigned short* xp    = (unsigned short*)(ws + XP_OFF);
  unsigned short* whb16 = (unsigned short*)(ws + WHB_OFF);
  unsigned short* hbuf  = (unsigned short*)(ws + HB_OFF);
  float* kst = (float*)(ws + K_OFF);
  float* kpw = (float*)(ws + KP_OFF);

  hipLaunchKernelGGL(init_kernel, dim3(2048), dim3(256), 0, stream,
                     Whf, Whb, whb16, (unsigned int*)hbuf, kst, kpw);

  // big input-projection GEMM: grid (N/64, M/128, dir)
  hipLaunchKernelGGL(xp_gemm_kernel, dim3(40, 256, 2), dim3(256), 0, stream,
                     x, Wxf, bxf, Wxb, bxb, xp);

  // sequential recurrence: one launch per timestep (graph replay amortizes)
  for (int s = 0; s < T_; ++s) {
    hipLaunchKernelGGL(lstm_step_kernel, dim3(32, 2), dim3(128), 0, stream,
                       xp, whb16, bhf, bhb, hbuf, kst, kpw, out, s);
  }
}
